// MambaWrapper_57432302682329
// MI455X (gfx1250) — matrix operand order repaired and checked by reading, not compiled, not run on hardware
//
#include <hip/hip_runtime.h>

// ---------------- problem constants ----------------
#define BN      4
#define LSEQ    2048
#define DMODEL  1024
#define DINNER  2048
#define DSTATE  16
#define DTRANK  64
#define MBL     (BN * LSEQ)          // 8192 rows

typedef __attribute__((ext_vector_type(16))) __bf16 v16bf;
typedef __attribute__((ext_vector_type(8)))  __bf16 v8bf;
typedef __attribute__((ext_vector_type(8)))  float  v8f;

// ---------------- fragment loaders (contiguous-k layouts, wave32) ----------------
// A 16x32 bf16 tile: lane = 16*h + r holds row (row0+r); element e maps to
// K = h*16 + e  ->  two contiguous 8-half (16B) chunks.
__device__ __forceinline__ v16bf load_a_frag(const __bf16* __restrict__ A, int lda,
                                             int row0, int k0, int lane) {
    int r  = lane & 15;
    int h  = lane >> 4;
    const __bf16* p = A + (long long)(row0 + r) * lda + k0 + h * 16;
    v8bf lo = *(const v8bf*)(p);
    v8bf hi = *(const v8bf*)(p + 8);
    v16bf a;
#pragma unroll
    for (int i = 0; i < 8; ++i) { a[i] = lo[i]; a[i + 8] = hi[i]; }
    return a;
}

// B 32x16 bf16 tile from weights stored [N,K] row-major: lane = 16*h + c holds
// column (col0+c), elements = K = h*16 + 0..15 (contiguous 32B).
// Column index is CLAMPED (not zero-filled): an OOB B column only corrupts its
// own output column, which the store guard discards -> unconditional loads,
// no exec-mask manipulation in the hot loop.
__device__ __forceinline__ v16bf load_b_frag(const __bf16* __restrict__ Bw, int ldb,
                                             int col0, int k0, int lane, int N) {
    int c = col0 + (lane & 15);
    c = (c < N) ? c : (N - 1);
    int h = lane >> 4;
    return *(const v16bf*)(Bw + (long long)c * ldb + k0 + h * 16);
}

__device__ __forceinline__ float fast_silu(float v) {
    return v / (1.0f + __expf(-v));
}

// ---------------- generic bf16 WMMA GEMM: C[M,N] = A[M,K] * Bw[N,K]^T ----------------
// block = 256 threads = 8 waves (2 rows x 4 cols of waves); each wave computes a
// 32x32 output tile (2 A-frags x 2 B-frags -> 4 WMMAs per k-step). Block tile 64x128.
// modes: 1 = plain store
//        2 = plain store + bf16 copy of cols<DTRANK into aux (ld 64)
//        3 = softplus(acc + bias[col]) store   (delta)
//        4 = out_proj epilogue: rev==0 -> C[row] = v ; rev==1 -> C[b*L+(L-1-l)] += v
__global__ __launch_bounds__(256) void wmma_gemm_bf16(
    const __bf16* __restrict__ A, const __bf16* __restrict__ Bw,
    int M, int N, int K,
    float* __restrict__ C, int ldc,
    int mode,
    const float* __restrict__ bias,
    __bf16* __restrict__ aux,
    int Lr, int rev)
{
    const int lane = threadIdx.x & 31;
    const int wv   = threadIdx.x >> 5;
    const int wr   = wv >> 2;            // 0..1
    const int wc   = wv & 3;             // 0..3
    const int row0 = blockIdx.y * 64 + wr * 32;
    const int col0 = blockIdx.x * 128 + wc * 32;
    if (row0 >= M) return;

    v8f c00 = {}, c01 = {}, c10 = {}, c11 = {};

    for (int k0 = 0; k0 < K; k0 += 32) {
        v16bf a0 = load_a_frag(A, K, row0,      k0, lane);
        v16bf a1 = load_a_frag(A, K, row0 + 16, k0, lane);
        v16bf b0 = load_b_frag(Bw, K, col0,      k0, lane, N);
        v16bf b1 = load_b_frag(Bw, K, col0 + 16, k0, lane, N);
        c00 = __builtin_amdgcn_wmma_f32_16x16x32_bf16(false, a0, false, b0, (short)0, c00, false, false);
        c01 = __builtin_amdgcn_wmma_f32_16x16x32_bf16(false, a0, false, b1, (short)0, c01, false, false);
        c10 = __builtin_amdgcn_wmma_f32_16x16x32_bf16(false, a1, false, b0, (short)0, c10, false, false);
        c11 = __builtin_amdgcn_wmma_f32_16x16x32_bf16(false, a1, false, b1, (short)0, c11, false, false);
    }

    // C/D layout: VGPR j -> row = row0 + t*16 + j + 8*h ; col = coltile + (lane&15)
    const int h = lane >> 4;
#pragma unroll
    for (int u = 0; u < 2; ++u) {
        const int col = col0 + u * 16 + (lane & 15);
        if (col >= N) continue;
#pragma unroll
        for (int t = 0; t < 2; ++t) {
            v8f acc;
            if (u == 0) acc = t ? c10 : c00;
            else        acc = t ? c11 : c01;
            int rbase = row0 + t * 16 + 8 * h;
#pragma unroll
            for (int j = 0; j < 8; ++j) {
                int row = rbase + j;
                float v = acc[j];
                if (mode == 3) {
                    v += bias[col];
                    v = (v > 20.0f) ? v : log1pf(__expf(v));
                    C[(long long)row * ldc + col] = v;
                } else if (mode == 4) {
                    if (rev) {
                        int b_ = row / Lr;
                        int l_ = row - b_ * Lr;
                        long long dst = (long long)(b_ * Lr + (Lr - 1 - l_)) * ldc + col;
                        C[dst] += v;
                    } else {
                        C[(long long)row * ldc + col] = v;
                    }
                } else {
                    C[(long long)row * ldc + col] = v;
                    if (mode == 2 && col < DTRANK)
                        aux[(long long)row * DTRANK + col] = (__bf16)v;
                }
            }
        }
    }
}

// ---------------- converts ----------------
__global__ void cvt_f32_bf16(const float* __restrict__ in, __bf16* __restrict__ out, long long n) {
    long long i = (long long)blockIdx.x * blockDim.x + threadIdx.x;
    long long stride = (long long)gridDim.x * blockDim.x;
    for (; i < n; i += stride) out[i] = (__bf16)in[i];
}

// hidden_states -> bf16 (forward order) and bf16 (time-reversed order)
__global__ void cvt_h_fwd_rev(const float* __restrict__ h,
                              __bf16* __restrict__ hbf, __bf16* __restrict__ hbf_r) {
    long long i = (long long)blockIdx.x * blockDim.x + threadIdx.x;
    long long stride = (long long)gridDim.x * blockDim.x;
    const long long n = (long long)MBL * DMODEL;
    for (; i < n; i += stride) {
        int k = (int)(i % DMODEL);
        long long bl = i / DMODEL;
        int l = (int)(bl % LSEQ);
        int b = (int)(bl / LSEQ);
        float v = h[i];
        hbf[i] = (__bf16)v;
        hbf_r[((long long)b * LSEQ + (LSEQ - 1 - l)) * DMODEL + k] = (__bf16)v;
    }
}

// ---------------- depthwise causal conv (D_CONV=4) + SiLU ----------------
// reads x = xz[:, 0:DINNER], writes bf16 silu(conv(x)) into xc
__global__ __launch_bounds__(256) void dwconv_silu(
    const float* __restrict__ xz, const float* __restrict__ w,
    const float* __restrict__ bias, __bf16* __restrict__ xc)
{
    const int d = blockIdx.x * 256 + threadIdx.x;   // 0..DINNER-1
    const int b = blockIdx.y;
    const float w0 = w[d * 4 + 0], w1 = w[d * 4 + 1], w2 = w[d * 4 + 2], w3 = w[d * 4 + 3];
    const float bb = bias[d];
    const float* px = xz + (long long)b * LSEQ * (2 * DINNER) + d;
    __bf16*      po = xc + (long long)b * LSEQ * DINNER + d;
    float x0 = 0.f, x1 = 0.f, x2 = 0.f;
    for (int l = 0; l < LSEQ; ++l) {
        float x3 = px[(long long)l * (2 * DINNER)];
        float a  = w0 * x0 + w1 * x1 + w2 * x2 + w3 * x3 + bb;
        po[(long long)l * DINNER] = (__bf16)fast_silu(a);
        x0 = x1; x1 = x2; x2 = x3;
    }
}

// ---------------- selective scan, fused D-term and silu(z) gate ----------------
__global__ __launch_bounds__(256) void selective_scan(
    const float* __restrict__ delta, const __bf16* __restrict__ xc,
    const float* __restrict__ xdbl, const float* __restrict__ xz,
    const float* __restrict__ A_log, const float* __restrict__ Dp,
    __bf16* __restrict__ ybf)
{
    const int d = blockIdx.x * 256 + threadIdx.x;   // 0..DINNER-1
    const int b = blockIdx.y;
    float Areg[DSTATE], s[DSTATE];
#pragma unroll
    for (int n = 0; n < DSTATE; ++n) {
        Areg[n] = -__expf(A_log[(long long)d * DSTATE + n]);
        s[n] = 0.f;
    }
    const float Dv = Dp[d];
    for (int l = 0; l < LSEQ; ++l) {
        const long long m = (long long)b * LSEQ + l;
        const float dt = delta[m * DINNER + d];
        const float xv = (float)xc[m * DINNER + d];
        const float* pb = xdbl + m * (DTRANK + 2 * DSTATE) + DTRANK;
        float bv[DSTATE], cv[DSTATE];
#pragma unroll
        for (int n = 0; n < DSTATE; ++n) { bv[n] = pb[n]; cv[n] = pb[DSTATE + n]; }
        const float dbx = dt * xv;
        float y = 0.f;
#pragma unroll
        for (int n = 0; n < DSTATE; ++n) {
            float dA = __expf(dt * Areg[n]);
            s[n] = dA * s[n] + dbx * bv[n];
            y += s[n] * cv[n];
        }
        const float zv = xz[m * (2 * DINNER) + DINNER + d];
        ybf[m * DINNER + d] = (__bf16)((y + xv * Dv) * fast_silu(zv));
    }
}

// ---------------- host ----------------
extern "C" void kernel_launch(void* const* d_in, const int* in_sizes, int n_in,
                              void* d_out, int out_size, void* d_ws, size_t ws_size,
                              hipStream_t stream) {
    const float* h = (const float*)d_in[0];
    struct DirIn {
        const float *in_proj, *conv_w, *conv_b, *x_proj, *dt_w, *dt_b, *A_log, *Dp, *out_proj;
    } dir[2];
    for (int i = 0; i < 2; ++i) {
        int o = 1 + i * 9;
        dir[i].in_proj  = (const float*)d_in[o + 0];
        dir[i].conv_w   = (const float*)d_in[o + 1];
        dir[i].conv_b   = (const float*)d_in[o + 2];
        dir[i].x_proj   = (const float*)d_in[o + 3];
        dir[i].dt_w     = (const float*)d_in[o + 4];
        dir[i].dt_b     = (const float*)d_in[o + 5];
        dir[i].A_log    = (const float*)d_in[o + 6];
        dir[i].Dp       = (const float*)d_in[o + 7];
        dir[i].out_proj = (const float*)d_in[o + 8];
    }

    size_t off = 0;
    auto alloc = [&](size_t bytes) -> void* {
        size_t a = (off + 255) & ~(size_t)255;
        off = a + bytes;
        return (void*)((char*)d_ws + a);
    };

    __bf16* hbf    = (__bf16*)alloc((size_t)MBL * DMODEL * 2);
    __bf16* hbf_r  = (__bf16*)alloc((size_t)MBL * DMODEL * 2);
    __bf16* w_in[2], *w_x[2], *w_dt[2], *w_out[2];
    for (int i = 0; i < 2; ++i) {
        w_in[i]  = (__bf16*)alloc((size_t)(2 * DINNER) * DMODEL * 2);
        w_x[i]   = (__bf16*)alloc((size_t)(DTRANK + 2 * DSTATE) * DINNER * 2);
        w_dt[i]  = (__bf16*)alloc((size_t)DINNER * DTRANK * 2);
        w_out[i] = (__bf16*)alloc((size_t)DMODEL * DINNER * 2);
    }
    float*  xz    = (float*) alloc((size_t)MBL * (2 * DINNER) * 4);
    __bf16* xconv = (__bf16*)alloc((size_t)MBL * DINNER * 2);
    float*  xdbl  = (float*) alloc((size_t)MBL * (DTRANK + 2 * DSTATE) * 4);
    __bf16* xdt   = (__bf16*)alloc((size_t)MBL * DTRANK * 2);
    float*  delta = (float*) alloc((size_t)MBL * DINNER * 4);
    __bf16* ybf   = (__bf16*)alloc((size_t)MBL * DINNER * 2);

    // --- one-time converts to bf16 ---
    cvt_h_fwd_rev<<<4096, 256, 0, stream>>>(h, hbf, hbf_r);
    for (int i = 0; i < 2; ++i) {
        cvt_f32_bf16<<<2048, 256, 0, stream>>>(dir[i].in_proj,  w_in[i],  (long long)(2 * DINNER) * DMODEL);
        cvt_f32_bf16<<<512,  256, 0, stream>>>(dir[i].x_proj,   w_x[i],   (long long)(DTRANK + 2 * DSTATE) * DINNER);
        cvt_f32_bf16<<<512,  256, 0, stream>>>(dir[i].dt_w,     w_dt[i],  (long long)DINNER * DTRANK);
        cvt_f32_bf16<<<2048, 256, 0, stream>>>(dir[i].out_proj, w_out[i], (long long)DMODEL * DINNER);
    }

    float* out = (float*)d_out;
    const int My = MBL / 64;   // 128

    for (int i = 0; i < 2; ++i) {
        const __bf16* Ah = (i == 0) ? hbf : hbf_r;

        // 1) in_proj: xz[MBL, 4096]
        wmma_gemm_bf16<<<dim3((2 * DINNER) / 128, My), 256, 0, stream>>>(
            Ah, w_in[i], MBL, 2 * DINNER, DMODEL, xz, 2 * DINNER,
            /*mode*/1, nullptr, nullptr, 0, 0);

        // 2) depthwise conv + silu -> xconv (bf16)
        dwconv_silu<<<dim3(DINNER / 256, BN), 256, 0, stream>>>(
            xz, dir[i].conv_w, dir[i].conv_b, xconv);

        // 3) x_proj: x_dbl[MBL, 96] (+ bf16 dt cols into xdt)
        wmma_gemm_bf16<<<dim3(1, My), 256, 0, stream>>>(
            xconv, w_x[i], MBL, DTRANK + 2 * DSTATE, DINNER, xdbl, DTRANK + 2 * DSTATE,
            /*mode*/2, nullptr, xdt, 0, 0);

        // 4) delta = softplus(xdt @ dt_w^T + dt_b)  [MBL, DINNER]
        wmma_gemm_bf16<<<dim3(DINNER / 128, My), 256, 0, stream>>>(
            xdt, w_dt[i], MBL, DINNER, DTRANK, delta, DINNER,
            /*mode*/3, dir[i].dt_b, nullptr, 0, 0);

        // 5) selective scan + D-term + silu(z) gate -> ybf (bf16)
        selective_scan<<<dim3(DINNER / 256, BN), 256, 0, stream>>>(
            delta, xconv, xdbl, xz, dir[i].A_log, dir[i].Dp, ybf);

        // 6) out_proj -> d_out (i==1: time-reversed accumulate)
        wmma_gemm_bf16<<<dim3(DMODEL / 128, My), 256, 0, stream>>>(
            ybf, w_out[i], MBL, DMODEL, DINNER, out, DMODEL,
            /*mode*/4, nullptr, nullptr, LSEQ, i);
    }
}
